// CARAFE_upsample_30880814858248
// MI455X (gfx1250) — matrix operand order repaired and checked by reading, not compiled, not run on hardware
//
#include <hip/hip_runtime.h>
#include <hip/hip_bf16.h>

typedef __attribute__((ext_vector_type(16))) __bf16 v16bf;
typedef __attribute__((ext_vector_type(8)))  float  v8f;

#define B_     2
#define H_     64
#define W_     64
#define HW_    4096          // 64*64
#define CIN_   256
#define CMID_  32
#define KUP_   5
#define K2_    25
#define NG_    4             // r*r sub-pixels
#define NOUTCH_ 100          // r*r*k*k encoder outputs

// workspace layout (bytes)
#define OFF_WCOMP  0u                  // 2*8*32*16 bf16   = 16 KB
#define OFF_WENC   16384u              // 7*25*32*16 bf16  = 179200 B
#define OFF_Y1     195584u             // B*HW*32 bf16     = 512 KB
#define OFF_MASK   1244160u            // B*HW*4*25 f32    = 3.277 MB
#define OFF_XT     4520960u            // B*HW*256 f32     = 8.389 MB

// k order this code packs for 16-bit A and B fragments (contiguous, not AMD's table order):
// lane l (row = l&15, half = l>>4), element e in v16bf:
//   K = e + half*16
__device__ __forceinline__ int kidx16(int e, int half) {
    return e + half * 16;
}

// ---------------------------------------------------------------------------
// Prep: pack w_comp [32,256] fp32 -> bf16 A fragments [mt(2)][kstep(8)][lane(32)][e(16)]
// ---------------------------------------------------------------------------
__global__ void pack_wcomp_kernel(const float* __restrict__ w_comp,
                                  __bf16* __restrict__ wcp) {
    int idx = blockIdx.x * 256 + threadIdx.x;   // 8192 total
    if (idx >= 2 * 8 * 32 * 16) return;
    int e     = idx & 15;
    int lane  = (idx >> 4) & 31;
    int kstep = (idx >> 9) & 7;
    int mt    = idx >> 12;
    int row   = mt * 16 + (lane & 15);
    int k     = kstep * 32 + kidx16(e, lane >> 4);
    wcp[idx] = (__bf16)w_comp[row * CIN_ + k];
}

// ---------------------------------------------------------------------------
// Prep: pack w_enc [100,32,5,5] fp32 -> bf16 A fragments [mt(7)][tap(25)][lane(32)][e(16)]
// K per tap = 32 (= C_MID)
// ---------------------------------------------------------------------------
__global__ void pack_wenc_kernel(const float* __restrict__ w_enc,
                                 __bf16* __restrict__ wep) {
    int idx = blockIdx.x * 256 + threadIdx.x;   // 89600 total
    if (idx >= 7 * K2_ * 32 * 16) return;
    int e    = idx & 15;
    int lane = (idx >> 4) & 31;
    int rest = idx >> 9;
    int tap  = rest % K2_;
    int mt   = rest / K2_;
    int o    = mt * 16 + (lane & 15);
    int c    = kidx16(e, lane >> 4);
    float v  = 0.f;
    if (o < NOUTCH_) v = w_enc[(o * CMID_ + c) * K2_ + tap];
    wep[idx] = (__bf16)v;
}

// ---------------------------------------------------------------------------
// Prep: transpose x [B,256,H,W] -> xt [B,H,W,256] (channels-last, coalesced writes)
// ---------------------------------------------------------------------------
__global__ void transpose_x_kernel(const float* __restrict__ x,
                                   float* __restrict__ xt) {
    int idx = blockIdx.x * 256 + threadIdx.x;   // 2,097,152 total
    int c  = idx & 255;
    int pw = idx >> 8;       // b*HW + h*W + w
    int w  = pw & 63;
    int h  = (pw >> 6) & 63;
    int b  = pw >> 12;
    xt[idx] = x[((b * CIN_ + c) * H_ + h) * W_ + w];
}

// ---------------------------------------------------------------------------
// Kernel 1: 1x1 compression conv as WMMA GEMM.
// y1[b][pix][m] (bf16, channels-last, m<32) = sum_c w_comp[m][c] * x[b][c][pix] + b_comp[m]
// Block: 256 threads (8 waves), 64 pixels; wave -> (mt = wv>>2, nt = wv&3).
// ---------------------------------------------------------------------------
__global__ void compress_gemm_kernel(const float* __restrict__ x,
                                     const float* __restrict__ b_comp,
                                     const v16bf* __restrict__ wcp,
                                     __bf16* __restrict__ y1) {
    __shared__ __bf16 Xlds[64 * CIN_];          // [pixel][c], 32 KB

    const int t    = threadIdx.x;
    const int lane = t & 31;
    const int wv   = t >> 5;
    const int b    = blockIdx.x >> 6;
    const int pix0 = (blockIdx.x & 63) * 64;

    const float* xb = x + (size_t)b * CIN_ * HW_;
    // stage 64 pixels x 256 channels, fp32 -> bf16
    #pragma unroll 4
    for (int it = 0; it < 64; ++it) {
        int e  = it * 256 + t;
        int c  = e >> 6;
        int pl = e & 63;
        Xlds[pl * CIN_ + c] = (__bf16)xb[(size_t)c * HW_ + pix0 + pl];
    }
    __syncthreads();

    const int mt = wv >> 2;                     // 0..1
    const int nt = wv & 3;                      // 0..3
    const int col  = (lane & 15);               // pixel within tile
    const int half = lane >> 4;
    const int pl   = nt * 16 + col;             // pixel within block
    const int c0   = half * 16;                 // B-frag: contiguous 16 channels

    v8f acc = {};
    #pragma unroll
    for (int kk = 0; kk < 8; ++kk) {
        v16bf a = wcp[(mt * 8 + kk) * 32 + lane];
        v16bf bmat = *(const v16bf*)&Xlds[pl * CIN_ + kk * 32 + c0];
        acc = __builtin_amdgcn_wmma_f32_16x16x32_bf16(
                  false, a, false, bmat, (short)0, acc, false, false);
    }

    // D layout: lane l, vgpr v -> row = v + (l>>4)*8, col = l&15
    const size_t pix = (size_t)pix0 + pl;
    #pragma unroll
    for (int v = 0; v < 8; ++v) {
        int m = mt * 16 + v + half * 8;         // channel 0..31
        y1[((size_t)b * HW_ + pix) * CMID_ + m] = (__bf16)(acc[v] + b_comp[m]);
    }
}

// ---------------------------------------------------------------------------
// Kernel 2: 5x5 encoder conv (WMMA, K-loop over 25 taps x K=32) + fused softmax.
// Block: 16 pixels (one row segment), 8 waves: waves 0..6 = M tiles (112 >= 100).
// ---------------------------------------------------------------------------
__global__ void encoder_softmax_kernel(const __bf16* __restrict__ y1,
                                       const float* __restrict__ b_enc,
                                       const v16bf* __restrict__ wep,
                                       float* __restrict__ mask) {
    __shared__ float score[112 * 16];           // [o][pix]

    const int t    = threadIdx.x;
    const int lane = t & 31;
    const int wv   = t >> 5;
    const int b    = blockIdx.x >> 8;           // 256 blocks per batch
    const int rem  = blockIdx.x & 255;
    const int h    = rem >> 2;
    const int w0   = (rem & 3) * 16;

    const int col  = lane & 15;
    const int half = lane >> 4;
    const int w    = w0 + col;
    const int c0   = half * 16;

    if (wv < 7) {
        const int mt = wv;
        v8f acc = {};
        #pragma unroll
        for (int tap = 0; tap < K2_; ++tap) {
            const int dy = tap / KUP_ - 2;
            const int dx = tap % KUP_ - 2;
            v16bf a = wep[(mt * K2_ + tap) * 32 + lane];
            const int sy = h + dy, sx = w + dx;
            v16bf bmat;
            if (sy >= 0 && sy < H_ && sx >= 0 && sx < W_) {
                bmat = *(const v16bf*)&y1[((size_t)b * HW_ + sy * W_ + sx) * CMID_ + c0];
            } else {
                bmat = (v16bf)(__bf16)0.0f;
            }
            acc = __builtin_amdgcn_wmma_f32_16x16x32_bf16(
                      false, a, false, bmat, (short)0, acc, false, false);
        }
        #pragma unroll
        for (int v = 0; v < 8; ++v) {
            int o = mt * 16 + v + half * 8;     // 0..111
            float bias = (o < NOUTCH_) ? b_enc[o] : 0.f;
            score[o * 16 + col] = acc[v] + bias;
        }
    }
    __syncthreads();

    // softmax over 25 taps per (pixel, sub-pixel g); channel = kk*4 + g
    if (t < 64) {
        const int pix = t & 15;
        const int g   = t >> 4;
        float vals[K2_];
        float mx = -3.0e38f;
        #pragma unroll
        for (int kk = 0; kk < K2_; ++kk) {
            float v = score[(kk * NG_ + g) * 16 + pix];
            vals[kk] = v;
            mx = fmaxf(mx, v);
        }
        float s = 0.f;
        #pragma unroll
        for (int kk = 0; kk < K2_; ++kk) {
            vals[kk] = __expf(vals[kk] - mx);
            s += vals[kk];
        }
        float inv = __frcp_rn(s);
        float* mp = mask + (((size_t)b * HW_ + h * W_ + w0 + pix) * NG_ + g) * K2_;
        #pragma unroll
        for (int kk = 0; kk < K2_; ++kk) mp[kk] = vals[kk] * inv;
    }
}

// ---------------------------------------------------------------------------
// Kernel 3: reassembly. One block per source pixel; thread = channel (coalesced).
// out[b][c][2h+r1][2w+r2] = sum_k xt[b][h+dy][w+dx][c] * mask[b][h][w][g][k]
// ---------------------------------------------------------------------------
__global__ void reassemble_kernel(const float* __restrict__ xt,
                                  const float* __restrict__ mask,
                                  float* __restrict__ out) {
    __shared__ float mL[NG_ * K2_];             // 100 floats

    const int c = threadIdx.x;                  // 0..255
    const int b = blockIdx.x >> 12;
    const int p = blockIdx.x & (HW_ - 1);
    const int h = p >> 6;
    const int w = p & 63;

    if (c < NG_ * K2_) mL[c] = mask[((size_t)b * HW_ + p) * (NG_ * K2_) + c];
    __syncthreads();

    float a0 = 0.f, a1 = 0.f, a2 = 0.f, a3 = 0.f;
    #pragma unroll
    for (int i = 0; i < KUP_; ++i) {
        const int sy = h + i - 2;
        #pragma unroll
        for (int j = 0; j < KUP_; ++j) {
            const int sx = w + j - 2;
            float v = 0.f;
            if (sy >= 0 && sy < H_ && sx >= 0 && sx < W_)
                v = xt[((size_t)b * HW_ + sy * W_ + sx) * CIN_ + c];
            const int kk = i * KUP_ + j;
            a0 = fmaf(v, mL[0 * K2_ + kk], a0);
            a1 = fmaf(v, mL[1 * K2_ + kk], a1);
            a2 = fmaf(v, mL[2 * K2_ + kk], a2);
            a3 = fmaf(v, mL[3 * K2_ + kk], a3);
        }
    }
    float* ob = out + (((size_t)b * CIN_ + c) * (2 * H_) + 2 * h) * (2 * W_) + 2 * w;
    ob[0]           = a0;   // (r1=0, r2=0)
    ob[1]           = a1;   // (r1=0, r2=1)
    ob[2 * W_]      = a2;   // (r1=1, r2=0)
    ob[2 * W_ + 1]  = a3;   // (r1=1, r2=1)
}

// ---------------------------------------------------------------------------
extern "C" void kernel_launch(void* const* d_in, const int* in_sizes, int n_in,
                              void* d_out, int out_size, void* d_ws, size_t ws_size,
                              hipStream_t stream) {
    const float* x      = (const float*)d_in[0];
    const float* w_comp = (const float*)d_in[1];
    const float* b_comp = (const float*)d_in[2];
    const float* w_enc  = (const float*)d_in[3];
    const float* b_enc  = (const float*)d_in[4];
    float* out = (float*)d_out;

    char* wsb = (char*)d_ws;
    __bf16* wcp  = (__bf16*)(wsb + OFF_WCOMP);
    __bf16* wep  = (__bf16*)(wsb + OFF_WENC);
    __bf16* y1   = (__bf16*)(wsb + OFF_Y1);
    float*  mask = (float*)(wsb + OFF_MASK);
    float*  xt   = (float*)(wsb + OFF_XT);

    pack_wcomp_kernel<<<32, 256, 0, stream>>>(w_comp, wcp);
    pack_wenc_kernel<<<350, 256, 0, stream>>>(w_enc, wep);
    transpose_x_kernel<<<(B_ * HW_ * CIN_) / 256, 256, 0, stream>>>(x, xt);
    compress_gemm_kernel<<<B_ * (HW_ / 64), 256, 0, stream>>>(
        x, b_comp, (const v16bf*)wcp, y1);
    encoder_softmax_kernel<<<B_ * (HW_ / 16), 256, 0, stream>>>(
        y1, b_enc, (const v16bf*)wep, mask);
    reassemble_kernel<<<B_ * HW_, 256, 0, stream>>>(xt, mask, out);
}
